// DeformableAttention3D_80144089743520
// MI455X (gfx1250) — matrix operand order repaired and checked by reading, not compiled, not run on hardware
//
#include <hip/hip_runtime.h>
#include <hip/hip_bf16.h>

typedef __attribute__((ext_vector_type(16))) _Float16 v16h;
typedef __attribute__((ext_vector_type(8)))  float    v8f;
typedef unsigned int u32x4 __attribute__((ext_vector_type(4)));
typedef int          i32x8 __attribute__((ext_vector_type(8)));
typedef int          i32x4 __attribute__((ext_vector_type(4)));

union VH16 { v16h v; _Float16 h[16]; };
union VF8  { v8f  v; float f[8]; };

// ---------------------------------------------------------------------------
// WMMA helper (CDNA5, wave32): D = A(16x32 f16) * B(32x16 f16) + C(16x16 f32)
// ---------------------------------------------------------------------------
static __device__ inline v8f wmma_f16(v16h a, v16h b, v8f c) {
  return __builtin_amdgcn_wmma_f32_16x16x32_f16(
      /*neg_a=*/false, a, /*neg_b=*/false, b,
      /*c_mod=*/(short)0, c, /*reuse_a=*/false, /*reuse_b=*/false);
}

// A-matrix 16x32 f16 fragment. Element A[m][k] at p[m*sm + k] (sk==1).
// lanes 0-15: M=lane, K=0..15 ; lanes 16-31: same M, K=16..31.
// Each lane reads one contiguous 16-half (32B) run.
static __device__ inline v16h frag_a(const _Float16* p, int sm) {
  const int lane = threadIdx.x & 31;
  const int m = lane & 15;
  const int hi = (lane >> 4) & 1;
  VH16 u;
#pragma unroll
  for (int e = 0; e < 16; ++e) {
    int k = hi * 16 + e;
    u.h[e] = p[m * sm + k];
  }
  return u.v;
}

// B-matrix 32x16 f16 fragment. Element B[k][n] at p[n*sn + k] (K contiguous).
// lanes 0-15 hold K=0..15 (N=lane), lanes 16-31 hold K=16..31.
// Each lane reads one contiguous 16-half (32B) run.
static __device__ inline v16h frag_b(const _Float16* p, int sn) {
  const int lane = threadIdx.x & 31;
  const int n = lane & 15;
  const int hi = (lane >> 4) & 1;
  VH16 u;
#pragma unroll
  for (int e = 0; e < 16; ++e) {
    int k = hi * 16 + e;
    u.h[e] = p[n * sn + k];
  }
  return u.v;
}

// ---------------------------------------------------------------------------
// TDM: async 2D f16 tile load (Global -> LDS), descriptor per ISA ch.8.
// tile_d0 elems/row, tile_d1 rows; stride0 = row stride in elements.
// ---------------------------------------------------------------------------
static __device__ inline void tdm_load_2d_f16(
    unsigned lds_off, const void* gptr,
    unsigned tile_d0, unsigned tile_d1,
    unsigned tensor_d0, unsigned tensor_d1, unsigned stride0) {
  unsigned long long ga = (unsigned long long)gptr;
  u32x4 g0;
  g0[0] = 1u;                                         // count=1, user D#
  g0[1] = lds_off;                                    // lds_addr (bytes)
  g0[2] = (unsigned)(ga & 0xffffffffu);               // global_addr[31:0]
  g0[3] = (unsigned)((ga >> 32) & 0x01ffffffu) | (2u << 30);  // addr[56:32]|type=2
  i32x8 g1;
  g1[0] = (int)(1u << 16);                            // data_size=1 (2 bytes)
  g1[1] = (int)((tensor_d0 & 0xffffu) << 16);         // abar_addr=0 | dim0.lo16
  g1[2] = (int)((tensor_d0 >> 16) | ((tensor_d1 & 0xffffu) << 16));
  g1[3] = (int)((tensor_d1 >> 16) | (tile_d0 << 16)); // dim1.hi16 | tile_dim0
  g1[4] = (int)(tile_d1 & 0xffffu);                   // tile_dim1 | tile_dim2=0
  g1[5] = (int)stride0;                               // dim0_stride[31:0]
  g1[6] = 0;                                          // stride hi | dim1_stride lo
  g1[7] = 0;
  i32x4 z4 = {0, 0, 0, 0};
#if defined(__clang_major__) && (__clang_major__ >= 23)
  i32x8 z8 = {0, 0, 0, 0, 0, 0, 0, 0};
  __builtin_amdgcn_tensor_load_to_lds(g0, g1, z4, z4, z8, 0);
#else
  __builtin_amdgcn_tensor_load_to_lds(g0, g1, z4, z4, 0);
#endif
}

// ---------------------------------------------------------------------------
// Problem constants
// ---------------------------------------------------------------------------
#define BS      2
#define HEADS   8
#define NQ      512          // 8^3 query tokens
#define NV      4096         // 16^3 kv tokens
#define DHEAD   64
#define QDIM    128
#define KVDIM   128
#define CPB     32
#define BH      (BS * HEADS) // == bs*groups == 16

// ---------------------------------------------------------------------------
// K1: q = gconv(query, Wq); offsets = tanh(Woff @ gelu(q*w_dw+b_dw)); vgrid
// ---------------------------------------------------------------------------
__global__ __launch_bounds__(256) void k_proj_q_off(
    const float* __restrict__ query, const float* __restrict__ Wq,
    const float* __restrict__ w_dw, const float* __restrict__ b_dw,
    const float* __restrict__ Woff,
    _Float16* __restrict__ Qh, float* __restrict__ vgrid) {
  const int b = blockIdx.x >> 3, g = blockIdx.x & 7, bg = blockIdx.x;
  __shared__ float Wqs[64 * 16];
  __shared__ float wdws[64], bdws[64], Woffs[3 * 64];
  const int tid = threadIdx.x;
  for (int i = tid; i < 64 * 16; i += 256) Wqs[i] = Wq[g * 64 * 16 + i];
  if (tid < 64) { wdws[tid] = w_dw[tid]; bdws[tid] = b_dw[tid]; }
  for (int i = tid; i < 192; i += 256) Woffs[i] = Woff[i];
  __syncthreads();

  for (int p = tid; p < NQ; p += 256) {
    float x[16];
    const float* qp = query + ((size_t)b * NQ + p) * QDIM + g * 16;
#pragma unroll
    for (int ci = 0; ci < 16; ++ci) x[ci] = qp[ci];
    float a0 = 0.f, a1 = 0.f, a2 = 0.f;
    _Float16* qo = Qh + ((size_t)bg * NQ + p) * DHEAD;
    for (int co = 0; co < 64; ++co) {
      float s = 0.f;
#pragma unroll
      for (int ci = 0; ci < 16; ++ci) s += Wqs[co * 16 + ci] * x[ci];
      qo[co] = (_Float16)(s * 0.125f);  // * dim_head^-0.5
      float t = s * wdws[co] + bdws[co];
      float ge = 0.5f * t * (1.0f + erff(t * 0.70710678118654752f));
      a0 += Woffs[co] * ge;
      a1 += Woffs[64 + co] * ge;
      a2 += Woffs[128 + co] * ge;
    }
    const int d = p >> 6, h = (p >> 3) & 7, w = p & 7;
    float* vg = vgrid + ((size_t)bg * NQ + p) * 3;
    vg[0] = ((float)d + tanhf(a0)) * (2.0f / 7.0f) - 1.0f;
    vg[1] = ((float)h + tanhf(a1)) * (2.0f / 7.0f) - 1.0f;
    vg[2] = ((float)w + tanhf(a2)) * (2.0f / 7.0f) - 1.0f;
  }
}

// ---------------------------------------------------------------------------
// K2: trilinear grid-sample of 4 volumes + Wk/Wv gconv.
// Kh stored row-major [bh][j][d]; V stored TRANSPOSED: Vt[bh][d][j].
// ---------------------------------------------------------------------------
__global__ __launch_bounds__(256) void k_sample_kv(
    const float* __restrict__ key, const float* __restrict__ value,
    const float* __restrict__ fkey, const float* __restrict__ fvalue,
    const float* __restrict__ Wk, const float* __restrict__ Wv,
    const float* __restrict__ vgrid,
    _Float16* __restrict__ Kh, _Float16* __restrict__ Vt) {
  const int b = blockIdx.x >> 3, g = blockIdx.x & 7, bg = blockIdx.x;
  __shared__ float Wks[64 * 32], Wvs[64 * 32];
  const int tid = threadIdx.x;
  for (int i = tid; i < 2048; i += 256) {
    Wks[i] = Wk[g * 2048 + i];
    Wvs[i] = Wv[g * 2048 + i];
  }
  __syncthreads();

  for (int p = tid; p < NQ; p += 256) {
    const float* vg = vgrid + ((size_t)bg * NQ + p) * 3;
    // torch grid_sample axis convention: comp0->W, comp1->H, comp2->D
    const float x = ((vg[0] + 1.f) * 16.f - 1.f) * 0.5f;
    const float y = ((vg[1] + 1.f) * 16.f - 1.f) * 0.5f;
    const float z = ((vg[2] + 1.f) * 16.f - 1.f) * 0.5f;
    const float x0 = floorf(x), y0 = floorf(y), z0 = floorf(z);
    float mk[32], mv[32];
#pragma unroll
    for (int c = 0; c < 32; ++c) { mk[c] = 0.f; mv[c] = 0.f; }
#pragma unroll
    for (int crn = 0; crn < 8; ++crn) {
      const float xi = x0 + (float)(crn & 1);
      const float yi = y0 + (float)((crn >> 1) & 1);
      const float zi = z0 + (float)((crn >> 2) & 1);
      if (xi < 0.f || xi > 15.f || yi < 0.f || yi > 15.f || zi < 0.f || zi > 15.f)
        continue;
      const float wgt =
          (1.f - fabsf(x - xi)) * (1.f - fabsf(y - yi)) * (1.f - fabsf(z - zi));
      const int lin = (((int)zi) * 16 + (int)yi) * 16 + (int)xi;
      const size_t base = ((size_t)b * NV + lin) * KVDIM + g * 16;
#pragma unroll
      for (int c = 0; c < 16; ++c) {
        mk[2 * c]     += wgt * key[base + c];
        mk[2 * c + 1] += wgt * fkey[base + c];
        mv[2 * c]     += wgt * value[base + c];
        mv[2 * c + 1] += wgt * fvalue[base + c];
      }
    }
    _Float16* ko = Kh + ((size_t)bg * NQ + p) * DHEAD;
    _Float16* vo = Vt + (size_t)bg * NQ * DHEAD + p;  // column p of Vt
    for (int co = 0; co < 64; ++co) {
      float sk = 0.f, sv = 0.f;
#pragma unroll
      for (int j = 0; j < 32; ++j) {
        sk += Wks[co * 32 + j] * mk[j];
        sv += Wvs[co * 32 + j] * mv[j];
      }
      ko[co] = (_Float16)sk;
      vo[(size_t)co * NQ] = (_Float16)sv;
    }
  }
}

// ---------------------------------------------------------------------------
// K3: CPB bias MLP (3->32->32->1), middle 32x32 layer via WMMA; writes S.
// grid: (8 j-tiles, 512 i, 16 bg), 64 threads (2 waves).
// ---------------------------------------------------------------------------
__global__ __launch_bounds__(64) void k_cpb_bias(
    const float* __restrict__ vgrid,
    const float* __restrict__ W1, const float* __restrict__ b1,
    const float* __restrict__ W2, const float* __restrict__ b2,
    const float* __restrict__ W3, const float* __restrict__ b3,
    float* __restrict__ S) {
  const int bg = blockIdx.z, i = blockIdx.y, j0 = blockIdx.x * 64;
  __shared__ __align__(16) _Float16 h1s[64 * 32];
  __shared__ __align__(16) _Float16 W2t[32 * 32];  // transposed: W2t[n][k]
  __shared__ float h2s[64 * 32];
  const int tid = threadIdx.x;
  for (int idx = tid; idx < 1024; idx += 64) {
    const int n = idx >> 5, kk = idx & 31;
    W2t[idx] = (_Float16)W2[kk * 32 + n];
  }

  const int d = i >> 6, h = (i >> 3) & 7, w = i & 7;
  const float gq0 = (float)d * (2.f / 7.f) - 1.f;
  const float gq1 = (float)h * (2.f / 7.f) - 1.f;
  const float gq2 = (float)w * (2.f / 7.f) - 1.f;

  const int j = j0 + tid;
  const float* vg = vgrid + ((size_t)bg * NQ + j) * 3;
  const float p0 = gq0 - vg[0], p1 = gq1 - vg[1], p2 = gq2 - vg[2];
  const float x0 = copysignf(log1pf(fabsf(p0)), p0);
  const float x1 = copysignf(log1pf(fabsf(p1)), p1);
  const float x2 = copysignf(log1pf(fabsf(p2)), p2);
#pragma unroll
  for (int c = 0; c < CPB; ++c) {
    float s = x0 * W1[c] + x1 * W1[32 + c] + x2 * W1[64 + c] + b1[c];
    h1s[tid * 32 + c] = (_Float16)fmaxf(s, 0.f);
  }
  __syncthreads();

  const int wv = tid >> 5;
  const int lane = tid & 31;
  const int n = lane & 15;
  const int mo = (lane & 16) ? 8 : 0;
#pragma unroll
  for (int mt = 0; mt < 2; ++mt) {
    v16h a = frag_a(h1s + (wv * 32 + mt * 16) * 32, 32);
#pragma unroll
    for (int nh = 0; nh < 2; ++nh) {
      v16h bf = frag_b(W2t + nh * 16 * 32, 32);
      v8f c = {};
      c = wmma_f16(a, bf, c);
      VF8 u; u.v = c;
#pragma unroll
      for (int r = 0; r < 8; ++r)
        h2s[(wv * 32 + mt * 16 + mo + r) * 32 + nh * 16 + n] = u.f[r];
    }
  }
  __syncthreads();

  float acc = b3[0];
#pragma unroll
  for (int c = 0; c < CPB; ++c)
    acc += fmaxf(h2s[tid * 32 + c] + b2[c], 0.f) * W3[c];
  S[((size_t)bg * NQ + i) * NQ + j] = acc;
}

// ---------------------------------------------------------------------------
// K4: S = Qh * Kh^T + S(bias). 64x64 C tile, 128 threads (4 waves).
// LDS tiles staged by the Tensor Data Mover (async, TENSORcnt).
// ---------------------------------------------------------------------------
__global__ __launch_bounds__(128) void k_gemm_qk(
    const _Float16* __restrict__ Qh, const _Float16* __restrict__ Kh,
    float* __restrict__ S) {
  const int bh = blockIdx.z, i0 = blockIdx.y * 64, j0 = blockIdx.x * 64;
  __shared__ __align__(16) _Float16 At[64 * 64];
  __shared__ __align__(16) _Float16 Bt[64 * 64];
  const int tid = threadIdx.x;

  if (tid < 32) {  // one wave drives the TDM
    tdm_load_2d_f16((unsigned)(uintptr_t)At,
                    Qh + (size_t)bh * NQ * DHEAD + (size_t)i0 * DHEAD,
                    /*tile*/ 64, 64, /*tensor*/ 64, NQ, /*stride0*/ 64);
    tdm_load_2d_f16((unsigned)(uintptr_t)Bt,
                    Kh + (size_t)bh * NQ * DHEAD + (size_t)j0 * DHEAD,
                    64, 64, 64, NQ, 64);
    __builtin_amdgcn_s_wait_tensorcnt(0);
  }
  __syncthreads();

  const int m0 = (tid >> 5) * 16;
  v8f acc[4] = {{}, {}, {}, {}};
#pragma unroll
  for (int kk = 0; kk < 64; kk += 32) {
    v16h a = frag_a(At + m0 * 64 + kk, 64);
#pragma unroll
    for (int nt = 0; nt < 4; ++nt) {
      // B[k][n] = Kh_tile[nt*16+n][kk+k]  (K contiguous per lane)
      v16h bf = frag_b(Bt + (nt * 16) * 64 + kk, 64);
      acc[nt] = wmma_f16(a, bf, acc[nt]);
    }
  }

  const int lane = tid & 31;
  const int n = lane & 15;
  const int mo = (lane & 16) ? 8 : 0;
  float* Sp = S + (size_t)bh * NQ * NQ;
#pragma unroll
  for (int nt = 0; nt < 4; ++nt) {
    VF8 u; u.v = acc[nt];
#pragma unroll
    for (int r = 0; r < 8; ++r) {
      size_t off = (size_t)(i0 + m0 + mo + r) * NQ + (j0 + nt * 16 + n);
      Sp[off] += u.f[r];  // S already holds CPB bias
    }
  }
}

// ---------------------------------------------------------------------------
// K5: row softmax of S, write P (f16).
// ---------------------------------------------------------------------------
__global__ __launch_bounds__(128) void k_softmax(
    const float* __restrict__ S, _Float16* __restrict__ P) {
  const size_t base = (size_t)blockIdx.x * NQ;
  const int tid = threadIdx.x;
  __shared__ float red[128];
  float v[4];
#pragma unroll
  for (int k = 0; k < 4; ++k) v[k] = S[base + tid + 128 * k];
  float m = fmaxf(fmaxf(v[0], v[1]), fmaxf(v[2], v[3]));
  red[tid] = m;
  __syncthreads();
  for (int s = 64; s > 0; s >>= 1) {
    if (tid < s) red[tid] = fmaxf(red[tid], red[tid + s]);
    __syncthreads();
  }
  m = red[0];
  __syncthreads();
  float e[4], sum = 0.f;
#pragma unroll
  for (int k = 0; k < 4; ++k) { e[k] = expf(v[k] - m); sum += e[k]; }
  red[tid] = sum;
  __syncthreads();
  for (int s = 64; s > 0; s >>= 1) {
    if (tid < s) red[tid] += red[tid + s];
    __syncthreads();
  }
  const float inv = 1.0f / red[0];
#pragma unroll
  for (int k = 0; k < 4; ++k)
    P[base + tid + 128 * k] = (_Float16)(e[k] * inv);
}

// ---------------------------------------------------------------------------
// K6: O = P * V. V is pre-transposed (Vt[bh][d][j]) so B-fragments are
// K-contiguous. TDM stages both tiles into one LDS buffer each k-step.
// ---------------------------------------------------------------------------
__global__ __launch_bounds__(128) void k_gemm_pv(
    const _Float16* __restrict__ P, const _Float16* __restrict__ Vt,
    float* __restrict__ O) {
  const int bh = blockIdx.y, i0 = blockIdx.x * 64;
  __shared__ __align__(16) _Float16 At[64 * 64];  // P tile  [row i][k]
  __shared__ __align__(16) _Float16 Bt[64 * 64];  // Vt tile [d][k]
  const int tid = threadIdx.x;
  const _Float16* Pp = P + (size_t)bh * NQ * NQ;
  const _Float16* Vp = Vt + (size_t)bh * NQ * DHEAD;  // [64][512]
  const int m0 = (tid >> 5) * 16;
  v8f acc[4] = {{}, {}, {}, {}};

  for (int kb = 0; kb < NQ; kb += 64) {
    __syncthreads();  // previous compute done before TDM rewrites LDS
    if (tid < 32) {
      tdm_load_2d_f16((unsigned)(uintptr_t)At, Pp + (size_t)(i0)*NQ + kb,
                      /*tile*/ 64, 64, /*tensor*/ NQ, NQ, /*stride0*/ NQ);
      tdm_load_2d_f16((unsigned)(uintptr_t)Bt, Vp + kb,
                      64, 64, NQ, DHEAD, NQ);
      __builtin_amdgcn_s_wait_tensorcnt(0);
    }
    __syncthreads();
#pragma unroll
    for (int kk = 0; kk < 64; kk += 32) {
      v16h a = frag_a(At + m0 * 64 + kk, 64);
#pragma unroll
      for (int nt = 0; nt < 4; ++nt) {
        // B[k][n] = Vt[nt*16+n][kb+kk+k]  (K contiguous per lane)
        v16h bf = frag_b(Bt + (nt * 16) * 64 + kk, 64);
        acc[nt] = wmma_f16(a, bf, acc[nt]);
      }
    }
  }

  const int lane = tid & 31;
  const int n = lane & 15;
  const int mo = (lane & 16) ? 8 : 0;
  float* Op = O + (size_t)bh * NQ * DHEAD;
#pragma unroll
  for (int nt = 0; nt < 4; ++nt) {
    VF8 u; u.v = acc[nt];
#pragma unroll
    for (int r = 0; r < 8; ++r)
      Op[(size_t)(i0 + m0 + mo + r) * DHEAD + nt * 16 + n] = u.f[r];
  }
}

// ---------------------------------------------------------------------------
// K7: out[b][p][cq] = bo[cq] + sum_{h,d} Wo[cq][h*64+d] * O[b*8+h][p][d]
// ---------------------------------------------------------------------------
__global__ __launch_bounds__(128) void k_out_proj(
    const float* __restrict__ O, const float* __restrict__ Wo,
    const float* __restrict__ bo, float* __restrict__ out) {
  const int b = blockIdx.x >> 9, p = blockIdx.x & 511;
  const int cq = threadIdx.x;
  float acc = bo[cq];
  const float* wrow = Wo + (size_t)cq * 512;
#pragma unroll
  for (int h = 0; h < HEADS; ++h) {
    const float* op = O + ((size_t)(b * HEADS + h) * NQ + p) * DHEAD;
#pragma unroll 8
    for (int d2 = 0; d2 < DHEAD; ++d2) acc += wrow[h * DHEAD + d2] * op[d2];
  }
  out[((size_t)b * NQ + p) * QDIM + cq] = acc;
}

// ---------------------------------------------------------------------------
extern "C" void kernel_launch(void* const* d_in, const int* in_sizes, int n_in,
                              void* d_out, int out_size, void* d_ws, size_t ws_size,
                              hipStream_t stream) {
  const float* query  = (const float*)d_in[0];
  const float* key    = (const float*)d_in[1];
  const float* value  = (const float*)d_in[2];
  const float* fkey   = (const float*)d_in[3];
  const float* fvalue = (const float*)d_in[4];
  const float* Wq     = (const float*)d_in[5];
  const float* w_dw   = (const float*)d_in[6];
  const float* b_dw   = (const float*)d_in[7];
  const float* Woff   = (const float*)d_in[8];
  const float* W1     = (const float*)d_in[9];
  const float* b1     = (const float*)d_in[10];
  const float* W2     = (const float*)d_in[11];
  const float* b2     = (const float*)d_in[12];
  const float* W3     = (const float*)d_in[13];
  const float* b3     = (const float*)d_in[14];
  const float* Wk     = (const float*)d_in[15];
  const float* Wv     = (const float*)d_in[16];
  const float* Wo     = (const float*)d_in[17];
  const float* bo     = (const float*)d_in[18];
  float* out = (float*)d_out;

  // Workspace carve-up (~29.2 MB)
  char* ws = (char*)d_ws;
  _Float16* Qh    = (_Float16*)(ws);                                   // 1 MB
  _Float16* Kh    = (_Float16*)(ws + (1u << 20));                      // 1 MB
  _Float16* Vt    = (_Float16*)(ws + (2u << 20));                      // 1 MB (transposed)
  float*    vgrid = (float*)   (ws + (3u << 20));                      // 96 KB
  float*    S     = (float*)   (ws + (3u << 20) + (128u << 10));       // 16 MB
  _Float16* P     = (_Float16*)(ws + (19u << 20) + (128u << 10));      // 8 MB
  float*    O     = (float*)   (ws + (27u << 20) + (128u << 10));      // 2 MB

  k_proj_q_off<<<BH, 256, 0, stream>>>(query, Wq, w_dw, b_dw, Woff, Qh, vgrid);
  k_sample_kv<<<BH, 256, 0, stream>>>(key, value, fkey, fvalue, Wk, Wv, vgrid, Kh, Vt);

  dim3 gbias(NQ / 64, NQ, BH);
  k_cpb_bias<<<gbias, 64, 0, stream>>>(vgrid, W1, b1, W2, b2, W3, b3, S);

  dim3 gqk(NQ / 64, NQ / 64, BH);
  k_gemm_qk<<<gqk, 128, 0, stream>>>(Qh, Kh, S);

  k_softmax<<<BH * NQ, 128, 0, stream>>>(S, P);

  dim3 gpv(NQ / 64, BH);
  k_gemm_pv<<<gpv, 128, 0, stream>>>(P, Vt, O);

  k_out_proj<<<BS * NQ, 128, 0, stream>>>(O, Wo, bo, out);
}
